// AdaptiveSoftmax_38122129719724
// MI455X (gfx1250) — compile-verified, matrix operand order checked by reading, not run on hardware
//
#include <hip/hip_runtime.h>
#include <hip/hip_bf16.h>

typedef __attribute__((ext_vector_type(16))) __bf16 v16bf;
typedef __attribute__((ext_vector_type(8)))  float  v8f;

union Frag { uint4 u[2]; v16bf v; };

constexpr int NROWS = 4096;
constexpr int NDIM  = 1024;
constexpr int C0 = 2000, C1 = 10000;
constexpr int HEAD = 2002, HEADP = 2016;   // padded to /16
constexpr int SZ0 = 8000,  SZ0P = 8000;
constexpr int SZ1 = 40257, SZ1P = 40272;
constexpr int D0 = 256, D1 = 64;
constexpr int SPLIT = 2;                   // vocab splits per 32-row group
constexpr int BLKS_PER_STAGE = (NROWS / 32) * SPLIT;   // 256
static_assert(BLKS_PER_STAGE == 256, "stage decode uses >>8");

// ---------- fp32 -> bf16 elementwise ----------
__global__ void cvt_bf16(const float* __restrict__ in, __bf16* __restrict__ out, int n) {
    int i = blockIdx.x * blockDim.x + threadIdx.x;
    if (i < n) out[i] = (__bf16)in[i];
}

// ---------- fp32 [K, NC] -> bf16 [NC, K] (transpose) ----------
__global__ void cvt_t_bf16(const float* __restrict__ in, __bf16* __restrict__ out, int K, int NC) {
    int i = blockIdx.x * blockDim.x + threadIdx.x;
    if (i < K * NC) {
        int k = i / NC, j = i % NC;
        out[(size_t)j * K + k] = (__bf16)in[i];
    }
}

// ---------- projections p0 = x@lin0, p1 = x@lin1 (bf16 out) ----------
__global__ __launch_bounds__(32) void proj_kernel(const __bf16* __restrict__ xb,
                                                  const __bf16* __restrict__ w0t,
                                                  const __bf16* __restrict__ w1t,
                                                  __bf16* __restrict__ p0,
                                                  __bf16* __restrict__ p1) {
    int tile = blockIdx.x;
    int rt = tile / 20, ct = tile % 20;
    int l = threadIdx.x, half = (l >= 16) ? 1 : 0, l16 = l & 15;

    const __bf16* B; __bf16* P; int pncols, cbase;
    if (ct < 16) { B = w0t; P = p0; pncols = D0; cbase = ct * 16; }
    else         { B = w1t; P = p1; pncols = D1; cbase = (ct - 16) * 16; }

    int arow = rt * 16 + l16;
    int bcol = cbase + l16;
    v8f acc = {};
    for (int k0 = 0; k0 < NDIM; k0 += 32) {
        int ka = k0 + half * 8;
        Frag A, Bf;
        A.u[0]  = *(const uint4*)(xb + (size_t)arow * NDIM + ka);
        A.u[1]  = *(const uint4*)(xb + (size_t)arow * NDIM + ka + 16);
        Bf.u[0] = *(const uint4*)(B  + (size_t)bcol * NDIM + ka);
        Bf.u[1] = *(const uint4*)(B  + (size_t)bcol * NDIM + ka + 16);
        acc = __builtin_amdgcn_wmma_f32_16x16x32_bf16(false, A.v, false, Bf.v,
                                                      (short)0, acc, false, false);
    }
    #pragma unroll
    for (int v = 0; v < 8; ++v) {
        int r = rt * 16 + v + 8 * half;
        P[(size_t)r * pncols + (cbase + l16)] = (__bf16)acc[v];
    }
}

struct RedBufs {
    float m[8][32];
    float s[8][32];
    float t[8][32];
    int   tgt[32];
};

// ---------- fused GEMM + online softmax partials (one stage) ----------
// Block = 256 threads (8 waves), owns 32 rows (2 WMMA row tiles) x one vocab split.
// Each B fragment feeds 2 WMMAs. Emits per-row (max, sumexp, target_logit) partials.
template<int K, int KPAD, int MODE>
__device__ __forceinline__ void lse_stage(const __bf16* __restrict__ A,
                                          const __bf16* __restrict__ W,
                                          const int* __restrict__ targets,
                                          float* __restrict__ pm,
                                          float* __restrict__ ps,
                                          float* __restrict__ pt,
                                          int ncols, int ntiles_pad, int bx,
                                          __bf16* a_lds, RedBufs& rb) {
    int rg  = bx / SPLIT;           // 32-row group
    int sp  = bx % SPLIT;           // vocab split
    int tid = threadIdx.x;
    int row0 = rg * 32;

    // stage A tile (32 x K bf16) into padded LDS
    for (int i = tid; i < 32 * (K / 8); i += 256) {
        int r = i / (K / 8), kc = (i % (K / 8)) * 8;
        *(uint4*)&a_lds[r * KPAD + kc] = *(const uint4*)(A + (size_t)(row0 + r) * K + kc);
    }
    if (tid < 32) {
        int t = targets[row0 + tid];
        int c;
        if (MODE == 0)      c = (t >= C0) ? ((t < C1) ? C0 : C0 + 1) : t;
        else if (MODE == 1) { c = t - C0; c = c < 0 ? 0 : (c > SZ0 - 1 ? SZ0 - 1 : c); }
        else                { c = t - C1; c = c < 0 ? 0 : (c > SZ1 - 1 ? SZ1 - 1 : c); }
        rb.tgt[tid] = c;
    }
    __syncthreads();

    int wave = tid >> 5, l = tid & 31, half = (l >= 16) ? 1 : 0, l16 = l & 15;

    float m[2][8], s[2][8], tl[2][8];
    int tgt[2][8];
    #pragma unroll
    for (int mt = 0; mt < 2; ++mt)
        #pragma unroll
        for (int v = 0; v < 8; ++v) {
            m[mt][v] = -1e30f; s[mt][v] = 0.f; tl[mt][v] = 0.f;
            tgt[mt][v] = rb.tgt[mt * 16 + v + 8 * half];
        }

    int gw = sp * 8 + wave;                    // 0..15 global wave over splits
    for (int nt = gw; nt < ntiles_pad; nt += SPLIT * 8) {
        int col = nt * 16 + l16;
        v8f acc0 = {}, acc1 = {};
        for (int k0 = 0; k0 < K; k0 += 32) {
            int ka = k0 + half * 8;
            Frag Bf, A0, A1;
            Bf.u[0] = *(const uint4*)(W + (size_t)col * K + ka);
            Bf.u[1] = *(const uint4*)(W + (size_t)col * K + ka + 16);
            A0.u[0] = *(const uint4*)&a_lds[l16 * KPAD + ka];
            A0.u[1] = *(const uint4*)&a_lds[l16 * KPAD + ka + 16];
            A1.u[0] = *(const uint4*)&a_lds[(16 + l16) * KPAD + ka];
            A1.u[1] = *(const uint4*)&a_lds[(16 + l16) * KPAD + ka + 16];
            acc0 = __builtin_amdgcn_wmma_f32_16x16x32_bf16(false, A0.v, false, Bf.v,
                                                           (short)0, acc0, false, false);
            acc1 = __builtin_amdgcn_wmma_f32_16x16x32_bf16(false, A1.v, false, Bf.v,
                                                           (short)0, acc1, false, false);
        }
        if (col < ncols) {
            #pragma unroll
            for (int v = 0; v < 8; ++v) {
                float val = acc0[v];
                float mn = fmaxf(m[0][v], val);
                s[0][v] = s[0][v] * __expf(m[0][v] - mn) + __expf(val - mn);
                m[0][v] = mn;
                if (col == tgt[0][v]) tl[0][v] += val;
            }
            #pragma unroll
            for (int v = 0; v < 8; ++v) {
                float val = acc1[v];
                float mn = fmaxf(m[1][v], val);
                s[1][v] = s[1][v] * __expf(m[1][v] - mn) + __expf(val - mn);
                m[1][v] = mn;
                if (col == tgt[1][v]) tl[1][v] += val;
            }
        }
    }

    // reduce across 16-lane halves (halves hold disjoint rows)
    #pragma unroll
    for (int mt = 0; mt < 2; ++mt)
        #pragma unroll
        for (int v = 0; v < 8; ++v)
            #pragma unroll
            for (int off = 1; off < 16; off <<= 1) {
                float mo = __shfl_xor(m[mt][v],  off, 32);
                float so = __shfl_xor(s[mt][v],  off, 32);
                float to = __shfl_xor(tl[mt][v], off, 32);
                float mn = fmaxf(m[mt][v], mo);
                s[mt][v] = s[mt][v] * __expf(m[mt][v] - mn) + so * __expf(mo - mn);
                m[mt][v] = mn;
                tl[mt][v] += to;
            }
    if (l16 == 0) {
        #pragma unroll
        for (int mt = 0; mt < 2; ++mt)
            #pragma unroll
            for (int v = 0; v < 8; ++v) {
                int r = mt * 16 + v + 8 * half;
                rb.m[wave][r] = m[mt][v];
                rb.s[wave][r] = s[mt][v];
                rb.t[wave][r] = tl[mt][v];
            }
    }
    __syncthreads();
    if (tid < 32) {
        float M = -1e30f;
        #pragma unroll
        for (int w = 0; w < 8; ++w) M = fmaxf(M, rb.m[w][tid]);
        float S = 0.f, T = 0.f;
        #pragma unroll
        for (int w = 0; w < 8; ++w) {
            S += rb.s[w][tid] * __expf(rb.m[w][tid] - M);
            T += rb.t[w][tid];
        }
        size_t idx = (size_t)sp * NROWS + row0 + tid;
        pm[idx] = M; ps[idx] = S; pt[idx] = T;
    }
}

// ---------- all three stages in one launch: overlap WMMA-bound head with
// TRANS/VALU-bound tail1 on the same WGPs ----------
__global__ __launch_bounds__(256) void lse_fused(const __bf16* __restrict__ xb,
                                                 const __bf16* __restrict__ wh,
                                                 const __bf16* __restrict__ we0,
                                                 const __bf16* __restrict__ we1,
                                                 const __bf16* __restrict__ p0,
                                                 const __bf16* __restrict__ p1,
                                                 const int* __restrict__ targets,
                                                 float* __restrict__ pm,
                                                 float* __restrict__ ps,
                                                 float* __restrict__ pt) {
    __shared__ __align__(16) __bf16 a_lds[32 * (NDIM + 8)];
    __shared__ RedBufs rb;
    int stage = blockIdx.x >> 8;
    int bx    = blockIdx.x & 255;
    if (stage == 0) {
        lse_stage<NDIM, NDIM + 8, 0>(xb, wh, targets,
                                     pm, ps, pt, HEAD, HEADP / 16, bx, a_lds, rb);
    } else if (stage == 1) {
        lse_stage<D0, D0 + 8, 1>(p0, we0, targets,
                                 pm + 1 * SPLIT * NROWS, ps + 1 * SPLIT * NROWS,
                                 pt + 1 * SPLIT * NROWS, SZ0, SZ0P / 16, bx, a_lds, rb);
    } else {
        lse_stage<D1, D1 + 8, 2>(p1, we1, targets,
                                 pm + 2 * SPLIT * NROWS, ps + 2 * SPLIT * NROWS,
                                 pt + 2 * SPLIT * NROWS, SZ1, SZ1P / 16, bx, a_lds, rb);
    }
}

// ---------- merge splits, apply masks, loss ----------
// partial layout: [stage][split][NROWS]
__global__ __launch_bounds__(1024) void combine_kernel(const float* __restrict__ pm,
                                                       const float* __restrict__ ps,
                                                       const float* __restrict__ pt,
                                                       const int* __restrict__ targets,
                                                       float* __restrict__ out) {
    __shared__ float red[1024];
    int tid = threadIdx.x;
    float acc = 0.f;
    for (int n = tid; n < NROWS; n += 1024) {
        float lp[3];
        #pragma unroll
        for (int st = 0; st < 3; ++st) {
            size_t b = (size_t)st * SPLIT * NROWS + n;
            float m0 = pm[b], m1 = pm[b + NROWS];
            float M = fmaxf(m0, m1);
            float S = ps[b] * __expf(m0 - M) + ps[b + NROWS] * __expf(m1 - M);
            float T = pt[b] + pt[b + NROWS];
            lp[st] = T - (M + __logf(S));
        }
        int t = targets[n];
        float v = lp[0];
        if (t >= C0 && t < C1) v += lp[1];
        if (t >= C1)           v += lp[2];
        out[n] = v;
        acc += -v;
    }
    red[tid] = acc;
    __syncthreads();
    for (int off = 512; off > 0; off >>= 1) {
        if (tid < off) red[tid] += red[tid + off];
        __syncthreads();
    }
    if (tid == 0) out[NROWS] = red[0] / (float)NROWS;
}

extern "C" void kernel_launch(void* const* d_in, const int* in_sizes, int n_in,
                              void* d_out, int out_size, void* d_ws, size_t ws_size,
                              hipStream_t stream) {
    (void)in_sizes; (void)n_in; (void)out_size; (void)ws_size;
    const float* inputs = (const float*)d_in[0];
    const int*   targets= (const int*)  d_in[1];
    const float* head_W = (const float*)d_in[2];
    const float* emb0   = (const float*)d_in[3];
    const float* lin0   = (const float*)d_in[4];
    const float* emb1   = (const float*)d_in[5];
    const float* lin1   = (const float*)d_in[6];
    float* out = (float*)d_out;

    char* ws = (char*)d_ws;
    size_t off = 0;
    auto take = [&](size_t bytes) -> void* {
        void* p = ws + off;
        off = (off + bytes + 255) & ~(size_t)255;
        return p;
    };
    __bf16* xb  = (__bf16*)take((size_t)NROWS * NDIM * 2);
    __bf16* wh  = (__bf16*)take((size_t)HEADP * NDIM * 2);
    __bf16* we0 = (__bf16*)take((size_t)SZ0P  * D0   * 2);
    __bf16* we1 = (__bf16*)take((size_t)SZ1P  * D1   * 2);
    __bf16* w0t = (__bf16*)take((size_t)D0    * NDIM * 2);
    __bf16* w1t = (__bf16*)take((size_t)D1    * NDIM * 2);
    __bf16* p0  = (__bf16*)take((size_t)NROWS * D0   * 2);
    __bf16* p1  = (__bf16*)take((size_t)NROWS * D1   * 2);
    float*  pm  = (float*) take((size_t)3 * SPLIT * NROWS * 4);
    float*  ps  = (float*) take((size_t)3 * SPLIT * NROWS * 4);
    float*  pt  = (float*) take((size_t)3 * SPLIT * NROWS * 4);

    auto blocks = [](int n) { return (n + 255) / 256; };
    cvt_bf16<<<blocks(NROWS * NDIM), 256, 0, stream>>>(inputs, xb, NROWS * NDIM);
    cvt_bf16<<<blocks(HEAD * NDIM),  256, 0, stream>>>(head_W, wh, HEAD * NDIM);
    cvt_bf16<<<blocks(SZ0 * D0),     256, 0, stream>>>(emb0, we0, SZ0 * D0);
    cvt_bf16<<<blocks(SZ1 * D1),     256, 0, stream>>>(emb1, we1, SZ1 * D1);
    cvt_t_bf16<<<blocks(NDIM * D0),  256, 0, stream>>>(lin0, w0t, NDIM, D0);
    cvt_t_bf16<<<blocks(NDIM * D1),  256, 0, stream>>>(lin1, w1t, NDIM, D1);

    proj_kernel<<<(NROWS / 16) * 20, 32, 0, stream>>>(xb, w0t, w1t, p0, p1);

    lse_fused<<<3 * BLKS_PER_STAGE, 256, 0, stream>>>(xb, wh, we0, we1, p0, p1,
                                                      targets, pm, ps, pt);

    combine_kernel<<<1, 1024, 0, stream>>>(pm, ps, pt, targets, out);
}
